// TrainEMMA_44306882625881
// MI455X (gfx1250) — matrix operand order repaired and checked by reading, not compiled, not run on hardware
//
#include <hip/hip_runtime.h>
#include <hip/hip_bf16.h>

// ---------------------------------------------------------------------------
// MI455X (gfx1250) implementation notes:
//  * wave32; WMMA 16x16x32 f16 (f32 accum) for all dense GEMMs
//  * softmax-weight trick removes the 43-GFLOP temb GEMMs -> pipeline is
//    HBM-bound on one 170MB read of temb (~7us @ 23.3 TB/s)
//  * conv input tensor (76800 f16 per batch) staged fully in LDS (150KB of
//    the 320KB/WGP), A-fragments read with ds_load_b128
// ---------------------------------------------------------------------------

typedef __attribute__((ext_vector_type(16))) _Float16 v16h;
typedef __attribute__((ext_vector_type(8)))  _Float16 v8h;
typedef __attribute__((ext_vector_type(8)))  float    v8f;

__device__ __forceinline__ v16h frag_combine(v8h lo, v8h hi) {
  v16h r;
#pragma unroll
  for (int i = 0; i < 8; ++i) { r[i] = lo[i]; r[i + 8] = hi[i]; }
  return r;
}

__device__ __forceinline__ float lrelu(float x) { return x > 0.f ? x : 0.01f * x; }

// ---------------------------------------------------------------------------
// Kernel 0: f32 -> f16 transpose: dst[c*R + r] = (f16) src[r*C + c]
// Makes every WMMA B-fragment two 16B contiguous loads (K-contiguous per col).
// ---------------------------------------------------------------------------
__global__ void k_transpose_f16(const float* __restrict__ src,
                                _Float16* __restrict__ dst, int R, int C) {
  int i = blockIdx.x * blockDim.x + threadIdx.x;
  if (i >= R * C) return;
  int r = i / C, c = i - r * C;
  dst[(size_t)c * R + r] = (_Float16)src[(size_t)r * C + c];
}

// ---------------------------------------------------------------------------
// Kernel 1: per (b,s): token scores (temb@Wks / @Wvs; bias cancels in softmax),
// 36-token softmax, weighted token sums -> Xk, Xv (f16, [1536][768]).
// One WG of 768 threads per (b,s); temb column kept in registers (read once).
// ---------------------------------------------------------------------------
#define TOKN 36
#define DMN  768
__global__ __launch_bounds__(768) void k_token_pool(
    const float* __restrict__ temb, const float* __restrict__ Wks,
    const float* __restrict__ Wvs, _Float16* __restrict__ Xk,
    _Float16* __restrict__ Xv) {
  __shared__ float sK[TOKN], sV[TOKN], wK[TOKN], wV[TOKN];
  const int bs_ = blockIdx.x;   // b*3 + s
  const int d = threadIdx.x;    // 0..767
  const float* tp = temb + (size_t)bs_ * TOKN * DMN + d;
  float col[TOKN];
#pragma unroll
  for (int t = 0; t < TOKN; ++t) col[t] = tp[(size_t)t * DMN];
  const float wks = Wks[d], wvs = Wvs[d];
  if (d < TOKN) { sK[d] = 0.f; sV[d] = 0.f; }
  __syncthreads();
  const int lane = threadIdx.x & 31;
#pragma unroll
  for (int t = 0; t < TOKN; ++t) {
    float pk = col[t] * wks, pv = col[t] * wvs;
#pragma unroll
    for (int m = 16; m >= 1; m >>= 1) {
      pk += __shfl_xor(pk, m, 32);
      pv += __shfl_xor(pv, m, 32);
    }
    if (lane == 0) { atomicAdd(&sK[t], pk); atomicAdd(&sV[t], pv); }
  }
  __syncthreads();
  if (threadIdx.x == 0) {  // 36-element softmax, trivially serial
    float mK = -1e30f, mV = -1e30f;
    for (int t = 0; t < TOKN; ++t) { mK = fmaxf(mK, sK[t]); mV = fmaxf(mV, sV[t]); }
    float dK = 0.f, dV = 0.f;
    for (int t = 0; t < TOKN; ++t) {
      wK[t] = expf(sK[t] - mK); dK += wK[t];
      wV[t] = expf(sV[t] - mV); dV += wV[t];
    }
    const float rK = 1.f / dK, rV = 1.f / dV;
    for (int t = 0; t < TOKN; ++t) { wK[t] *= rK; wV[t] *= rV; }
  }
  __syncthreads();
  float xk = 0.f, xv = 0.f;
#pragma unroll
  for (int t = 0; t < TOKN; ++t) { xk += wK[t] * col[t]; xv += wV[t] * col[t]; }
  Xk[(size_t)bs_ * DMN + d] = (_Float16)xk;
  Xv[(size_t)bs_ * DMN + d] = (_Float16)xv;
}

// ---------------------------------------------------------------------------
// Kernel 2: k/v GEMM: X (1536x768 f16) @ W^T (256x768 f16) + bias -> 1536x256 f32
// One wave per 16x16 output tile, 24x v_wmma_f32_16x16x32_f16.
// ---------------------------------------------------------------------------
__global__ __launch_bounds__(32) void k_kv_gemm(
    const _Float16* __restrict__ Xk, const _Float16* __restrict__ Xv,
    const _Float16* __restrict__ WkT, const _Float16* __restrict__ WvT,
    const float* __restrict__ bk, const float* __restrict__ bv,
    float* __restrict__ Kmat, float* __restrict__ Vmat) {
  const int mt = blockIdx.x;     // 0..95
  const int nt = blockIdx.y;     // 0..15
  const int which = blockIdx.z;  // 0 = K, 1 = V
  const _Float16* X = which ? Xv : Xk;
  const _Float16* WT = which ? WvT : WkT;
  const float* bias = which ? bv : bk;
  float* Out = which ? Vmat : Kmat;
  const int lane = threadIdx.x, l15 = lane & 15, kh = lane >> 4;
  const _Float16* arow = X + (size_t)(mt * 16 + l15) * 768;
  const _Float16* brow = WT + (size_t)(nt * 16 + l15) * 768;
  v8f acc = {};
  for (int ks = 0; ks < 24; ++ks) {
    const int kb = ks * 32;
    v8h alo = *(const v8h*)(arow + kb + kh * 8);
    v8h ahi = *(const v8h*)(arow + kb + 16 + kh * 8);
    v8h blo = *(const v8h*)(brow + kb + kh * 8);
    v8h bhi = *(const v8h*)(brow + kb + 16 + kh * 8);
    acc = __builtin_amdgcn_wmma_f32_16x16x32_f16(
        false, frag_combine(alo, ahi), false, frag_combine(blo, bhi),
        (short)0, acc, false, false);
  }
  const int n = nt * 16 + l15;
  const float bb = bias[n];
#pragma unroll
  for (int r = 0; r < 8; ++r) {
    const int m = mt * 16 + r + kh * 8;
    Out[(size_t)m * 256 + n] = acc[r] + bb;
  }
}

// ---------------------------------------------------------------------------
// Kernel 3+4 fused (one WG per batch b, 256 threads):
//   - sprite_emb + v[b] staged in LDS; ke[i][s] = sprite_emb[i]·k[b,s]
//   - per-cell masked attention softmax; st = (att + avatar)/2 built as f16
//     directly in LDS (76800 elems = 150KB: CDNA5 320KB/WGP LDS)
//   - 2x2 VALID conv as 81x64x3072 WMMA GEMM, A-fragments from LDS
//     (ds_load_b128), B from pre-transposed f16 weights; lrelu; write Y in
//     transposed-flat (f*81 + p) order for the head GEMMs.
// ---------------------------------------------------------------------------
__global__ __launch_bounds__(256) void k_attn_conv(
    const int* __restrict__ state, const float* __restrict__ sprite,
    const float* __restrict__ Kmat, const float* __restrict__ Vmat,
    const _Float16* __restrict__ convT, const float* __restrict__ conv_b,
    _Float16* __restrict__ Y) {
  extern __shared__ char smem[];
  _Float16* sst = (_Float16*)smem;               // 76800 f16 (flat st buffer)
  float* ssp = (float*)(smem + 153600);          // 25*256 sprite_emb
  float* svl = ssp + 25 * 256;                   // 3*256   v[b]
  float* ske = svl + 3 * 256;                    // 75      ke[i*3+s]
  int* sfl = (int*)(ske + 76);                   // 25      nonzero-row flags
  const int b = blockIdx.x;
  const int tid = threadIdx.x;

  for (int i = tid; i < 25 * 256; i += 256) ssp[i] = sprite[i];
  for (int i = tid; i < 3 * 256; i += 256) svl[i] = Vmat[(size_t)b * 768 + i];
  __syncthreads();
  if (tid < 25) {
    float s = 0.f;
    for (int e = 0; e < 256; ++e) s += ssp[tid * 256 + e];
    sfl[tid] = (s != 0.0f) ? 1 : 0;
  }
  if (tid < 75) {
    const int i = tid / 3, s = tid - 3 * (tid / 3);
    const float* kr = Kmat + ((size_t)b * 3 + s) * 256;
    float acc = 0.f;
    for (int e = 0; e < 256; ++e) acc += ssp[i * 256 + e] * kr[e];
    ske[tid] = acc;
  }
  __syncthreads();

  // ---- Phase C: build st (flat, t-major scramble matches reference reshape)
  const int wave = tid >> 5, lane = tid & 31;
  for (int x = wave; x < 300; x += 8) {
    const int* sp = state + ((size_t)b * 300 + x) * 4;
    const int o0 = sp[0], o1 = sp[1], o2 = sp[2], av = sp[3];
    const int cnt = sfl[o0] + sfl[o1] + sfl[o2];
    const float inv = 1.0f / (float)(cnt > 1 ? cnt : 1) / 16.0f;  // /cnt /sqrt(EMB)
    float kq[3];
#pragma unroll
    for (int s = 0; s < 3; ++s)  // ke[0][s]==0 since sprite row 0 is zero
      kq[s] = (ske[o0 * 3 + s] + ske[o1 * 3 + s] + ske[o2 * 3 + s]) * inv;
    const float mx = fmaxf(kq[0], fmaxf(kq[1], kq[2]));
    const float e0 = expf(kq[0] - mx), e1 = expf(kq[1] - mx), e2 = expf(kq[2] - mx);
    const float rden = 1.0f / (e0 + e1 + e2);
    const float third = 1.0f / 3.0f;
    const float w0 = (kq[0] != 0.f) ? e0 * rden * third : 0.f;
    const float w1 = (kq[1] != 0.f) ? e1 * rden * third : 0.f;
    const float w2 = (kq[2] != 0.f) ? e2 * rden * third : 0.f;
#pragma unroll
    for (int u = 0; u < 8; ++u) {
      const int e = lane * 8 + u;
      const float att = w0 * svl[e] + w1 * svl[256 + e] + w2 * svl[512 + e];
      sst[x * 256 + e] = (_Float16)(0.5f * (att + ssp[av * 256 + e]));
    }
  }
  __syncthreads();

  // ---- Phase D: conv as WMMA GEMM (M=81 pad 96, N=64, K=3072)
  for (int tile = wave; tile < 24; tile += 8) {
    const int mt = tile >> 2, nt = tile & 3;
    const int l15 = lane & 15, kh = lane >> 4;
    const int p = mt * 16 + l15;
    const int pi = p / 9, pj = p - pi * 9;
    const bool valid = (p < 81);
    const _Float16* brow = convT + (size_t)(nt * 16 + l15) * 3072;
    v8f acc = {};
    for (int ks = 0; ks < 96; ++ks) {
      const int dblock = ks / 24;            // which (di,dj) tap
      const int di = dblock >> 1, dj = dblock & 1;
      const int coff = (ks - dblock * 24) * 32;
      v16h a = {};
      if (valid) {
        const _Float16* ap = sst + ((pi + di) * 10 + (pj + dj)) * 768 + coff + kh * 8;
        a = frag_combine(*(const v8h*)(ap), *(const v8h*)(ap + 16));
      }
      const _Float16* bp = brow + ks * 32 + kh * 8;
      v16h bf = frag_combine(*(const v8h*)(bp), *(const v8h*)(bp + 16));
      acc = __builtin_amdgcn_wmma_f32_16x16x32_f16(false, a, false, bf,
                                                   (short)0, acc, false, false);
    }
    const int f = nt * 16 + l15;
    const float cb = conv_b[f];
#pragma unroll
    for (int r = 0; r < 8; ++r) {
      const int pr = mt * 16 + r + kh * 8;
      if (pr < 81)
        Y[(size_t)b * 5184 + f * 81 + pr] = (_Float16)lrelu(acc[r] + cb);
    }
  }
}

// ---------------------------------------------------------------------------
// Kernel 5: head GEMMs: Y (512x5184 f16) @ a1T/v1T (128x5184 f16) + b, lrelu
// ---------------------------------------------------------------------------
__global__ __launch_bounds__(32) void k_head_gemm(
    const _Float16* __restrict__ Y, const _Float16* __restrict__ a1T,
    const _Float16* __restrict__ v1T, const float* __restrict__ a1b,
    const float* __restrict__ v1b, float* __restrict__ Ha,
    float* __restrict__ Hv) {
  const int mt = blockIdx.x;     // 0..31
  const int nt = blockIdx.y;     // 0..7
  const int which = blockIdx.z;  // 0 = actor, 1 = critic
  const _Float16* WT = which ? v1T : a1T;
  const float* bias = which ? v1b : a1b;
  float* Out = which ? Hv : Ha;
  const int lane = threadIdx.x, l15 = lane & 15, kh = lane >> 4;
  const _Float16* arow = Y + (size_t)(mt * 16 + l15) * 5184;
  const _Float16* brow = WT + (size_t)(nt * 16 + l15) * 5184;
  v8f acc = {};
  for (int ks = 0; ks < 162; ++ks) {
    const int kb = ks * 32;
    v8h alo = *(const v8h*)(arow + kb + kh * 8);
    v8h ahi = *(const v8h*)(arow + kb + 16 + kh * 8);
    v8h blo = *(const v8h*)(brow + kb + kh * 8);
    v8h bhi = *(const v8h*)(brow + kb + 16 + kh * 8);
    acc = __builtin_amdgcn_wmma_f32_16x16x32_f16(
        false, frag_combine(alo, ahi), false, frag_combine(blo, bhi),
        (short)0, acc, false, false);
  }
  const int n = nt * 16 + l15;
  const float bb = bias[n];
#pragma unroll
  for (int r = 0; r < 8; ++r) {
    const int m = mt * 16 + r + kh * 8;
    Out[(size_t)m * 128 + n] = lrelu(acc[r] + bb);
  }
}

// ---------------------------------------------------------------------------
// Kernel 6: tiny 128x128 layers, logits/log_softmax/entropy + value.
// out[0:512]=action_logprobs, out[512:1024]=state_value, out[1024:1536]=entropy
// ---------------------------------------------------------------------------
__global__ __launch_bounds__(128) void k_final(
    const float* __restrict__ Ha, const float* __restrict__ Hv,
    const float* __restrict__ a2, const float* __restrict__ a2b,
    const float* __restrict__ a3, const float* __restrict__ a3b,
    const float* __restrict__ v2, const float* __restrict__ v2b,
    const float* __restrict__ v3, const float* __restrict__ v3b,
    const int* __restrict__ action, float* __restrict__ out) {
  __shared__ float h1a[128], h1v[128], h2a[128], h2v[128];
  const int b = blockIdx.x, tid = threadIdx.x;
  h1a[tid] = Ha[b * 128 + tid];
  h1v[tid] = Hv[b * 128 + tid];
  __syncthreads();
  float sa = 0.f, sv = 0.f;
  for (int k = 0; k < 128; ++k) {
    sa += h1a[k] * a2[k * 128 + tid];
    sv += h1v[k] * v2[k * 128 + tid];
  }
  h2a[tid] = lrelu(sa + a2b[tid]);
  h2v[tid] = lrelu(sv + v2b[tid]);
  __syncthreads();
  if (tid < 5) {
    float l = a3b[tid];
    for (int k = 0; k < 128; ++k) l += h2a[k] * a3[k * 5 + tid];
    h1a[tid] = l;  // reuse as logits scratch (safe: all h1a reads done)
  }
  if (tid == 32) {
    float s = v3b[0];
    for (int k = 0; k < 128; ++k) s += h2v[k] * v3[k];
    out[512 + b] = s;
  }
  __syncthreads();
  if (tid == 0) {
    float m = h1a[0];
    for (int j = 1; j < 5; ++j) m = fmaxf(m, h1a[j]);
    float den = 0.f;
    for (int j = 0; j < 5; ++j) den += expf(h1a[j] - m);
    const float lse = m + logf(den);
    float ent = 0.f;
    for (int j = 0; j < 5; ++j) {
      const float lp = h1a[j] - lse;
      ent -= expf(lp) * lp;
    }
    out[b] = h1a[action[b]] - lse;
    out[1024 + b] = ent;
  }
}

// ---------------------------------------------------------------------------
extern "C" void kernel_launch(void* const* d_in, const int* in_sizes, int n_in,
                              void* d_out, int out_size, void* d_ws,
                              size_t ws_size, hipStream_t stream) {
  const int* state    = (const int*)d_in[0];
  const int* action   = (const int*)d_in[1];
  const float* temb   = (const float*)d_in[2];
  const float* sprite = (const float*)d_in[3];
  const float* conv_w = (const float*)d_in[4];
  const float* conv_b = (const float*)d_in[5];
  const float* Wk  = (const float*)d_in[6];
  const float* bk  = (const float*)d_in[7];
  const float* Wks = (const float*)d_in[8];
  // d_in[9] = bks: softmax is shift-invariant -> unused
  const float* Wv  = (const float*)d_in[10];
  const float* bv  = (const float*)d_in[11];
  const float* Wvs = (const float*)d_in[12];
  // d_in[13] = bvs: unused (shift-invariant)
  const float* a1  = (const float*)d_in[14];
  const float* a1b = (const float*)d_in[15];
  const float* a2  = (const float*)d_in[16];
  const float* a2b = (const float*)d_in[17];
  const float* a3  = (const float*)d_in[18];
  const float* a3b = (const float*)d_in[19];
  const float* v1  = (const float*)d_in[20];
  const float* v1b = (const float*)d_in[21];
  const float* v2  = (const float*)d_in[22];
  const float* v2b = (const float*)d_in[23];
  const float* v3  = (const float*)d_in[24];
  const float* v3b = (const float*)d_in[25];
  float* out = (float*)d_out;
  (void)in_sizes; (void)n_in; (void)out_size; (void)ws_size;

  char* ws = (char*)d_ws;
  size_t off = 0;
  auto alloc = [&](size_t bytes) -> char* {
    char* p = ws + off;
    off += (bytes + 255) & ~(size_t)255;
    return p;
  };
  _Float16* WkT   = (_Float16*)alloc(768 * 256 * 2);
  _Float16* WvT   = (_Float16*)alloc(768 * 256 * 2);
  _Float16* convT = (_Float16*)alloc(3072 * 64 * 2);
  _Float16* a1T   = (_Float16*)alloc(5184 * 128 * 2);
  _Float16* v1T   = (_Float16*)alloc(5184 * 128 * 2);
  _Float16* Xk    = (_Float16*)alloc(1536 * 768 * 2);
  _Float16* Xv    = (_Float16*)alloc(1536 * 768 * 2);
  float* Kmat     = (float*)alloc(1536 * 256 * 4);
  float* Vmat     = (float*)alloc(1536 * 256 * 4);
  _Float16* Y     = (_Float16*)alloc((size_t)512 * 5184 * 2);
  float* Ha       = (float*)alloc(512 * 128 * 4);
  float* Hv       = (float*)alloc(512 * 128 * 4);

  // 0: weight conversion / transpose to f16
  int n = 768 * 256;
  k_transpose_f16<<<(n + 255) / 256, 256, 0, stream>>>(Wk, WkT, 768, 256);
  k_transpose_f16<<<(n + 255) / 256, 256, 0, stream>>>(Wv, WvT, 768, 256);
  n = 3072 * 64;
  k_transpose_f16<<<(n + 255) / 256, 256, 0, stream>>>(conv_w, convT, 3072, 64);
  n = 5184 * 128;
  k_transpose_f16<<<(n + 255) / 256, 256, 0, stream>>>(a1, a1T, 5184, 128);
  k_transpose_f16<<<(n + 255) / 256, 256, 0, stream>>>(v1, v1T, 5184, 128);

  // 1: token pooling (HBM-bound: single 170MB read of temb)
  k_token_pool<<<1536, 768, 0, stream>>>(temb, Wks, Wvs, Xk, Xv);

  // 2: k/v GEMM via WMMA
  k_kv_gemm<<<dim3(96, 16, 2), 32, 0, stream>>>(Xk, Xv, WkT, WvT, bk, bv, Kmat, Vmat);

  // 3+4: attention + conv fused, st tensor lives in 150KB LDS
  const size_t smem = 153600 + (25 * 256 + 3 * 256 + 76) * 4 + 25 * 4 + 28;
  (void)hipFuncSetAttribute((const void*)k_attn_conv,
                            hipFuncAttributeMaxDynamicSharedMemorySize,
                            (int)smem);
  k_attn_conv<<<512, 256, smem, stream>>>(state, sprite, Kmat, Vmat, convT,
                                          conv_b, Y);

  // 5: head GEMMs via WMMA
  k_head_gemm<<<dim3(32, 8, 2), 32, 0, stream>>>(Y, a1T, v1T, a1b, v1b, Ha, Hv);

  // 6: tiny layers + log_softmax / entropy / value
  k_final<<<512, 128, 0, stream>>>(Ha, Hv, a2, a2b, a3, a3b, v2, v2b, v3, v3b,
                                   action, out);
}
